// Self_Attention_Adj_3676492006027
// MI455X (gfx1250) — matrix operand order repaired and checked by reading, not compiled, not run on hardware
//
#include <hip/hip_runtime.h>

typedef _Float16 h16;
typedef __attribute__((ext_vector_type(16))) _Float16 v16h;
typedef __attribute__((ext_vector_type(8)))  _Float16 v8h;
typedef __attribute__((ext_vector_type(8)))  float    v8f;

#define B_  8
#define F_  512
#define N_  2048
#define A_  256
#define NEG_SLOPE 0.01f

#define LDS_K      32
#define LDS_STRIDE 40   // 32 halves + 8 pad -> 80B rows, every chunk 16B aligned
#define KSTEPS     (F_ / LDS_K)   // 16

// ---------------------------------------------------------------------------
// Fragment loaders (f16, 16x16x32 WMMA; ISA 7.12.2 layouts, wave32)
//
// A-matrix 16x32: lanes 0-15 : halves 0..7 = K0..7,  halves 8..15 = K16..23
//                 lanes16-31 : halves 0..7 = K8..15, halves 8..15 = K24..31
// row_base points at element (row, K=0) of a [row][k] layout.
__device__ __forceinline__ v16h load_a_frag(const h16* row_base, int lane) {
    const h16* p = row_base + 8 * (lane >> 4);
    v8h lo = *(const v8h*)(p);        // K block at +0
    v8h hi = *(const v8h*)(p + 16);   // K block at +16
    return __builtin_shufflevector(lo, hi, 0,1,2,3,4,5,6,7,8,9,10,11,12,13,14,15);
}
// B-matrix 32x16: lanes 0-15 halves 0..7 = K0..7, halves 8..15 = K16..23 ; lanes 16-31 = K8..15, K24..31
// col_base points at element (col, K=0) of a [col][k] layout.
__device__ __forceinline__ v16h load_b_frag(const h16* col_base, int lane) {
    const h16* p = col_base + 8 * (lane >> 4);
    v8h lo = *(const v8h*)(p);
    v8h hi = *(const v8h*)(p + 16);
    return __builtin_shufflevector(lo, hi, 0,1,2,3,4,5,6,7,8,9,10,11,12,13,14,15);
}

// ---------------------------------------------------------------------------
// Kernel 1: Q = LeakyReLU(x^T Wq), K = LeakyReLU(x^T Wk)  -> f16 intermediates
// Combined GEMM over [Wq | Wk] (512 cols). Block tile 128x128, 8 waves, each
// wave 2x4 16x16 tiles, K-steps of 32, DOUBLE-BUFFERED LDS (ping-pong):
// per step: issue next tile's global loads -> WMMA on current tile (loads in
// flight) -> ds-store next tile -> one barrier. 2 x 20KB LDS of 320KB/WGP.
__global__ __launch_bounds__(256) void proj_qk_kernel(
    const float* __restrict__ x, const float* __restrict__ Wq,
    const float* __restrict__ Wk, h16* __restrict__ Qh, h16* __restrict__ Kh)
{
    __shared__ __align__(16) h16 As[2][128 * LDS_STRIDE];
    __shared__ __align__(16) h16 Bs[2][128 * LDS_STRIDE];

    const int b  = blockIdx.z;
    const int n0 = blockIdx.x * 128;
    const int c0 = blockIdx.y * 128;               // column in combined 512-wide W
    const float* W   = (c0 < A_) ? Wq : Wk;
    h16*         Out = (c0 < A_) ? Qh : Kh;
    const int cbase  = c0 & (A_ - 1);

    const int tid  = threadIdx.x;
    const int lane = tid & 31;
    const int wid  = tid >> 5;
    const int waveRow = wid >> 1;                  // 0..3 -> rows 32*waveRow
    const int waveCol = wid &  1;                  // 0..1 -> cols 64*waveCol

    const float* xb = x + (size_t)b * F_ * N_;

    // Per-thread staging coordinates: 4 float4 chunks, flat = kk*32 + q
    int kks[4], qs[4];
    #pragma unroll
    for (int it = 0; it < 4; ++it) {
        const int flat = tid + it * 256;           // 0..1023
        kks[it] = flat >> 5;                       // k within tile (0..31)
        qs[it]  = flat & 31;                       // float4 index along row/col dim
    }

    v8f acc[2][4] = {};
    float4 xv[4], wv[4];

    // ---- prologue: fetch + stage K-step 0 into buffer 0 ----
    #pragma unroll
    for (int it = 0; it < 4; ++it) {
        xv[it] = *(const float4*)(xb + (size_t)kks[it] * N_ + n0 + qs[it] * 4);
        wv[it] = *(const float4*)(W  + (size_t)kks[it] * A_ + cbase + qs[it] * 4);
    }
    #pragma unroll
    for (int it = 0; it < 4; ++it) {
        h16* d = &As[0][(qs[it] * 4) * LDS_STRIDE + kks[it]];
        d[0 * LDS_STRIDE] = (h16)xv[it].x; d[1 * LDS_STRIDE] = (h16)xv[it].y;
        d[2 * LDS_STRIDE] = (h16)xv[it].z; d[3 * LDS_STRIDE] = (h16)xv[it].w;
        h16* e = &Bs[0][(qs[it] * 4) * LDS_STRIDE + kks[it]];
        e[0 * LDS_STRIDE] = (h16)wv[it].x; e[1 * LDS_STRIDE] = (h16)wv[it].y;
        e[2 * LDS_STRIDE] = (h16)wv[it].z; e[3 * LDS_STRIDE] = (h16)wv[it].w;
    }
    __syncthreads();

    for (int step = 0; step < KSTEPS; ++step) {
        const int  cur     = step & 1;
        const bool hasNext = (step + 1) < KSTEPS;

        // ---- issue next tile's global loads (overlap with WMMA below) ----
        if (hasNext) {
            const int f0n = (step + 1) * LDS_K;
            #pragma unroll
            for (int it = 0; it < 4; ++it) {
                xv[it] = *(const float4*)(xb + (size_t)(f0n + kks[it]) * N_ + n0 + qs[it] * 4);
                wv[it] = *(const float4*)(W  + (size_t)(f0n + kks[it]) * A_ + cbase + qs[it] * 4);
            }
        }

        // ---- compute on current buffer ----
        v16h afrag[2], bfrag[4];
        #pragma unroll
        for (int rt = 0; rt < 2; ++rt)
            afrag[rt] = load_a_frag(
                &As[cur][(32 * waveRow + 16 * rt + (lane & 15)) * LDS_STRIDE], lane);
        #pragma unroll
        for (int ct = 0; ct < 4; ++ct)
            bfrag[ct] = load_b_frag(
                &Bs[cur][(64 * waveCol + 16 * ct + (lane & 15)) * LDS_STRIDE], lane);
        #pragma unroll
        for (int rt = 0; rt < 2; ++rt)
            #pragma unroll
            for (int ct = 0; ct < 4; ++ct)
                acc[rt][ct] = __builtin_amdgcn_wmma_f32_16x16x32_f16(
                    false, afrag[rt], false, bfrag[ct],
                    (short)0, acc[rt][ct], false, false);

        // ---- stage next tile into the other buffer, single barrier ----
        if (hasNext) {
            const int nxt = cur ^ 1;
            #pragma unroll
            for (int it = 0; it < 4; ++it) {
                h16* d = &As[nxt][(qs[it] * 4) * LDS_STRIDE + kks[it]];
                d[0 * LDS_STRIDE] = (h16)xv[it].x; d[1 * LDS_STRIDE] = (h16)xv[it].y;
                d[2 * LDS_STRIDE] = (h16)xv[it].z; d[3 * LDS_STRIDE] = (h16)xv[it].w;
                h16* e = &Bs[nxt][(qs[it] * 4) * LDS_STRIDE + kks[it]];
                e[0 * LDS_STRIDE] = (h16)wv[it].x; e[1 * LDS_STRIDE] = (h16)wv[it].y;
                e[2 * LDS_STRIDE] = (h16)wv[it].z; e[3 * LDS_STRIDE] = (h16)wv[it].w;
            }
            __syncthreads();
        }
    }

    // ---- epilogue: LeakyReLU + f16 convert + store ----
    #pragma unroll
    for (int rt = 0; rt < 2; ++rt) {
        #pragma unroll
        for (int ct = 0; ct < 4; ++ct) {
            const int col = cbase + 64 * waveCol + 16 * ct + (lane & 15);
            #pragma unroll
            for (int i = 0; i < 8; ++i) {
                // C/D layout: VGPR i, lanes0-15 -> M=i ; lanes16-31 -> M=8+i
                const int row = n0 + 32 * waveRow + 16 * rt + i + 8 * (lane >> 4);
                float v = acc[rt][ct][i];
                v = (v >= 0.f) ? v : NEG_SLOPE * v;
                Out[((size_t)b * N_ + row) * A_ + col] = (h16)v;
            }
        }
    }
}

// ---------------------------------------------------------------------------
// Kernel 2: S[b] = Q[b] * K[b]^T  (2048x2048, K-dim = 256, raw f32 scores)
// Q/K are f16 [b][n][a] row-major -> both A and B fragments are contiguous in
// the K(=a) dimension, so load fragments directly from global (operands are
// 16MB total: fully L2-resident on the 192MB L2). Prefetch next K-block.
__global__ __launch_bounds__(256) void scores_kernel(
    const h16* __restrict__ Qh, const h16* __restrict__ Kh, float* __restrict__ S)
{
    const int b  = blockIdx.z;
    const int n0 = blockIdx.x * 128;
    const int m0 = blockIdx.y * 128;

    const int tid  = threadIdx.x;
    const int lane = tid & 31;
    const int wid  = tid >> 5;
    const int waveRow = wid >> 1;
    const int waveCol = wid &  1;

    const h16* Qb = Qh + (size_t)b * N_ * A_;
    const h16* Kb = Kh + (size_t)b * N_ * A_;

    v8f acc[2][4] = {};

    #pragma unroll
    for (int a0 = 0; a0 < A_; a0 += 32) {
        v16h afrag[2], bfrag[4];
        #pragma unroll
        for (int rt = 0; rt < 2; ++rt) {
            const int row = n0 + 32 * waveRow + 16 * rt + (lane & 15);
            const h16* p = Qb + (size_t)row * A_ + a0;
            if (a0 + 32 < A_) __builtin_prefetch(p + 32, 0, 3);  // global_prefetch_b8
            afrag[rt] = load_a_frag(p, lane);
        }
        #pragma unroll
        for (int ct = 0; ct < 4; ++ct) {
            const int col = m0 + 64 * waveCol + 16 * ct + (lane & 15);
            const h16* p = Kb + (size_t)col * A_ + a0;
            if (a0 + 32 < A_) __builtin_prefetch(p + 32, 0, 3);
            bfrag[ct] = load_b_frag(p, lane);
        }
        #pragma unroll
        for (int rt = 0; rt < 2; ++rt)
            #pragma unroll
            for (int ct = 0; ct < 4; ++ct)
                acc[rt][ct] = __builtin_amdgcn_wmma_f32_16x16x32_f16(
                    false, afrag[rt], false, bfrag[ct],
                    (short)0, acc[rt][ct], false, false);
    }

    float* Sb = S + (size_t)b * N_ * N_;
    #pragma unroll
    for (int rt = 0; rt < 2; ++rt) {
        #pragma unroll
        for (int ct = 0; ct < 4; ++ct) {
            const int col = m0 + 64 * waveCol + 16 * ct + (lane & 15);
            #pragma unroll
            for (int i = 0; i < 8; ++i) {
                const int row = n0 + 32 * waveRow + 16 * rt + i + 8 * (lane >> 4);
                Sb[(size_t)row * N_ + col] = acc[rt][ct][i];
            }
        }
    }
}

// ---------------------------------------------------------------------------
// Kernel 3: softmax over axis 1 (per column m of each [N][N] matrix), in place.
// One block = one batch x 256 columns; lane t owns column m0+t, loops rows ->
// consecutive lanes touch consecutive addresses (fully coalesced). Online
// max/sum pass, then normalize pass: 2 reads + 1 write of the score tensor.
// Unroll-by-4 for memory-level parallelism on the HBM-bound stream.
__global__ __launch_bounds__(256) void softmax_col_kernel(float* __restrict__ S)
{
    const int b = blockIdx.y;
    const int m = blockIdx.x * 256 + threadIdx.x;
    float* Sb = S + (size_t)b * N_ * N_;

    float mx  = -3.0e38f;
    float sum = 0.f;
    #pragma unroll 4
    for (int n = 0; n < N_; ++n) {
        float v  = Sb[(size_t)n * N_ + m];
        float nm = fmaxf(mx, v);
        sum = sum * __expf(mx - nm) + __expf(v - nm);
        mx  = nm;
    }
    const float inv = 1.f / sum;
    #pragma unroll 4
    for (int n = 0; n < N_; ++n) {
        const size_t idx = (size_t)n * N_ + m;
        Sb[idx] = __expf(Sb[idx] - mx) * inv;
    }
}

// ---------------------------------------------------------------------------
extern "C" void kernel_launch(void* const* d_in, const int* in_sizes, int n_in,
                              void* d_out, int out_size, void* d_ws, size_t ws_size,
                              hipStream_t stream) {
    (void)in_sizes; (void)n_in; (void)out_size; (void)ws_size;

    const float* x  = (const float*)d_in[0];   // [8, 512, 2048]
    const float* Wq = (const float*)d_in[1];   // [512, 256]
    const float* Wk = (const float*)d_in[2];   // [512, 256]
    float* S = (float*)d_out;                  // [8, 2048, 2048]

    h16* Qh = (h16*)d_ws;                      // [8, 2048, 256] f16 (8 MB)
    h16* Kh = Qh + (size_t)B_ * N_ * A_;       // [8, 2048, 256] f16 (8 MB)

    dim3 blk(256);
    proj_qk_kernel    <<<dim3(N_ / 128, 512 / 128, B_), blk, 0, stream>>>(x, Wq, Wk, Qh, Kh);
    scores_kernel     <<<dim3(N_ / 128, N_ / 128, B_), blk, 0, stream>>>(Qh, Kh, S);
    softmax_col_kernel<<<dim3(N_ / 256, B_), blk, 0, stream>>>(S);
}
